// BidirectionalCrossPatchAttention_16209206575185
// MI455X (gfx1250) — compile-verified, matrix operand order checked by reading, not run on hardware
//
#include <hip/hip_runtime.h>
#include <hip/hip_bf16.h>

// ---------------------------------------------------------------------------
// BidirectionalCrossPatchAttention for MI455X (gfx1250, wave32, WMMA bf16)
//   B=4, K=2048, D=768, H=12, HD=64
// Pipeline: cvt(f32->bf16) -> QKV gemms (WMMA) -> flash attention (WMMA)
//           -> output projection (WMMA, f32 out)
// ---------------------------------------------------------------------------

typedef unsigned short u16;
typedef __attribute__((ext_vector_type(16))) __bf16 bf16x16;
typedef __attribute__((ext_vector_type(8)))  float  v8f;

#define DEVINL __device__ __forceinline__

static constexpr int Bb = 4, Kk = 2048, Dd = 768, Hh = 12, HDc = 64;
static constexpr int Mm = Bb * Kk;          // 8192 rows of x
static constexpr int LDS_STR = 40;          // ushorts per 32-wide staged row (+8 pad)
static constexpr int KS_STR  = 72;          // ushorts per 64-wide staged row (+8 pad)

// round-to-nearest-even f32 -> bf16 (bit trick; no __bf16 arithmetic needed)
DEVINL u16 f2bf(float f) {
  unsigned u = __builtin_bit_cast(unsigned, f);
  u += 0x7FFFu + ((u >> 16) & 1u);
  return (u16)(u >> 16);
}

struct FragBF {
  union { bf16x16 v; uint4 q[2]; };
};

// Load a 16-lane-wide WMMA operand fragment (A or B role) from a row-major
// tile: lane (lane&15) owns row; elements 0..7 at k = 8*half, 8..15 at
// k = 16 + 8*half (CDNA5 16-bit 16x32 layout). Two b128 loads per lane.
DEVINL void load_frag(FragBF& f, const u16* rowp, int half) {
  f.q[0] = *(const uint4*)(rowp + (half << 3));
  f.q[1] = *(const uint4*)(rowp + 16 + (half << 3));
}

DEVINL v8f wmma_bf16(const FragBF& a, const FragBF& b, v8f c) {
  return __builtin_amdgcn_wmma_f32_16x16x32_bf16(
      /*neg_a=*/false, a.v, /*neg_b=*/false, b.v,
      /*c_mod=*/(short)0, c, /*reuse_a=*/false, /*reuse_b=*/false);
}

// --- DPP16 16-lane reductions (pure VALU; no LDS/bpermute traffic) ---------
// Butterfly: quad_perm xor1, quad_perm xor2, row_half_mirror (xor7 pairs the
// two quads of an 8-group), row_mirror (xor15 pairs the two 8-groups of the
// 16-lane DPP row). EXEC is all-ones here (WMMA requirement), lanes 0..15 and
// 16..31 reduce independently = exactly the two accumulator halves.
template <int CTRL>
DEVINL float dpp_movf(float x) {
  int r = __builtin_amdgcn_update_dpp(
      0, __builtin_bit_cast(int, x), CTRL, 0xF, 0xF, true);
  return __builtin_bit_cast(float, r);
}
DEVINL float red16_max(float v) {
  v = fmaxf(v, dpp_movf<0x0B1>(v));  // quad_perm [1,0,3,2]
  v = fmaxf(v, dpp_movf<0x04E>(v));  // quad_perm [2,3,0,1]
  v = fmaxf(v, dpp_movf<0x141>(v));  // row_half_mirror
  v = fmaxf(v, dpp_movf<0x140>(v));  // row_mirror
  return v;
}
DEVINL float red16_sum(float v) {
  v += dpp_movf<0x0B1>(v);
  v += dpp_movf<0x04E>(v);
  v += dpp_movf<0x141>(v);
  v += dpp_movf<0x140>(v);
  return v;
}

// ---------------------------------------------------------------------------
// f32 -> bf16 elementwise conversion, 4 elements / iteration (n % 4 == 0)
// ---------------------------------------------------------------------------
__global__ void cvt_f32_bf16_kernel(const float* __restrict__ src,
                                    u16* __restrict__ dst, int n4) {
  int i = blockIdx.x * blockDim.x + threadIdx.x;
  int stride = gridDim.x * blockDim.x;
  for (; i < n4; i += stride) {
    float4 f = ((const float4*)src)[i];
    union { ushort4 s; u16 u[4]; } o;
    o.u[0] = f2bf(f.x); o.u[1] = f2bf(f.y);
    o.u[2] = f2bf(f.z); o.u[3] = f2bf(f.w);
    ((ushort4*)dst)[i] = o.s;
  }
}

// ---------------------------------------------------------------------------
// y = Xb @ Wb^T + bias   (Xb: [Md,Kd] bf16, Wb: [Nd,Kd] bf16)
// block = 256 threads (8 waves); block tile 128x64; wave tile 32x32 (2x2 WMMA)
// ---------------------------------------------------------------------------
template <bool F32OUT>
__global__ void __launch_bounds__(256)
gemm_bias_kernel(const u16* __restrict__ Xb, const u16* __restrict__ Wb,
                 const float* __restrict__ bias, u16* __restrict__ Ybf,
                 float* __restrict__ Yf, int Md, int Nd, int Kd) {
  __shared__ u16 Xs[128 * LDS_STR];
  __shared__ u16 Ws[64 * LDS_STR];

  const int tid = threadIdx.x;
  const int lane = tid & 31, wave = tid >> 5;
  const int half = lane >> 4, n16 = lane & 15;

  const int tileR = blockIdx.x * 128;
  const int tileC = blockIdx.y * 64;
  const int wrow = (wave >> 1) * 32;   // 0..96
  const int wcol = (wave & 1) * 32;    // 0 or 32

  v8f acc[2][2] = {};

  for (int k0 = 0; k0 < Kd; k0 += 32) {
    __syncthreads();
    // stage X tile 128x32 (b128 per thread-task)
    for (int t = tid; t < 128 * 4; t += 256) {
      int r = t >> 2, c = (t & 3) * 8;
      const u16* gp = &Xb[(size_t)(tileR + r) * Kd + k0 + c];
      *(uint4*)&Xs[r * LDS_STR + c] = *(const uint4*)gp;
      if (k0 + 32 < Kd) __builtin_prefetch(gp + 32, 0, 0);
    }
    // stage W tile 64x32 (exactly 256 tasks)
    {
      int r = tid >> 2, c = (tid & 3) * 8;
      *(uint4*)&Ws[r * LDS_STR + c] =
          *(const uint4*)&Wb[(size_t)(tileC + r) * Kd + k0 + c];
    }
    __syncthreads();

    FragBF a0, a1, b0, b1;
    load_frag(a0, &Xs[(wrow + n16) * LDS_STR], half);
    load_frag(a1, &Xs[(wrow + 16 + n16) * LDS_STR], half);
    load_frag(b0, &Ws[(wcol + n16) * LDS_STR], half);
    load_frag(b1, &Ws[(wcol + 16 + n16) * LDS_STR], half);

    acc[0][0] = wmma_bf16(a0, b0, acc[0][0]);
    acc[0][1] = wmma_bf16(a0, b1, acc[0][1]);
    acc[1][0] = wmma_bf16(a1, b0, acc[1][0]);
    acc[1][1] = wmma_bf16(a1, b1, acc[1][1]);
  }

  // epilogue: D[m][n] -> lane holds col n=lane&15, rows m = 8*half + r
  for (int rf = 0; rf < 2; ++rf) {
    for (int cf = 0; cf < 2; ++cf) {
      int col = tileC + wcol + cf * 16 + n16;
      float bv = bias[col];
      for (int r = 0; r < 8; ++r) {
        int row = tileR + wrow + rf * 16 + half * 8 + r;
        float y = acc[rf][cf][r] + bv;
        if (F32OUT) Yf[(size_t)row * Nd + col] = y;
        else        Ybf[(size_t)row * Nd + col] = f2bf(y);
      }
    }
  }
}

// ---------------------------------------------------------------------------
// Flash attention with bidirectional context mask.
// grid = (K/64, H, B); block = 128 (4 waves); wave owns 16 query rows.
// Q,Kb,Vb,Ob are bf16 [B,K,D]; head h occupies columns h*64 .. h*64+63.
// allowed(i,j) = !ctx[i] || ctx[j];   masked -> -1e30 (matches reference NEG)
// Softmax runs in the exp2 domain: scale*log2(e) folded into the score scale.
// ---------------------------------------------------------------------------
__global__ void __launch_bounds__(128)
flash_attn_kernel(const u16* __restrict__ Qb, const u16* __restrict__ Kbuf,
                  const u16* __restrict__ Vbuf,
                  const unsigned char* __restrict__ ctx,
                  u16* __restrict__ Ob) {
  __shared__ u16 Plds[4 * 16 * LDS_STR];  // per-wave 16x32 P tile (bf16)
  __shared__ u16 Ks[32 * KS_STR];         // shared K tile  [key][d]  (row major)
  __shared__ u16 Vt[64 * LDS_STR];        // shared V tile transposed [d][key]

  const int tid = threadIdx.x;
  const int lane = tid & 31, wave = tid >> 5;
  const int half = lane >> 4, n16 = lane & 15;

  const int qt = blockIdx.x, h = blockIdx.y, b = blockIdx.z;
  const int q0 = qt * 64 + wave * 16;
  const size_t baseBK = (size_t)b * Kk * Dd + (size_t)h * HDc;

  // resident Q fragments (HD=64 -> two k=32 fragments)
  FragBF qf[2];
  {
    const u16* qrow = Qb + baseBK + (size_t)(q0 + n16) * Dd;
    load_frag(qf[0], qrow, half);
    load_frag(qf[1], qrow + 32, half);
  }

  // context bits for the 8 query rows this lane's accumulators cover
  unsigned rowctx = 0;
  for (int r = 0; r < 8; ++r)
    rowctx |= (unsigned)(ctx[(size_t)b * Kk + q0 + half * 8 + r] != 0) << r;

  v8f o[4] = {};
  v8f m8, l8;
  for (int r = 0; r < 8; ++r) { m8[r] = -3.0e38f; l8[r] = 0.0f; }

  const float sc2 = 0.125f * 1.44269504088896f;  // (1/sqrt(64)) * log2(e)
  u16* myP = &Plds[wave * 16 * LDS_STR];

  for (int j0 = 0; j0 < Kk; j0 += 32) {
    __syncthreads();  // protect Ks/Vt from previous iteration's readers
    // stage K tile 32x64 row-major: Ks[j][d] = K[b, j0+j, h*64+d]
    for (int t = tid; t < 32 * 8; t += 128) {
      int jj = t >> 3, c = (t & 7) * 8;
      const u16* kp = &Kbuf[baseBK + (size_t)(j0 + jj) * Dd + c];
      *(uint4*)&Ks[jj * KS_STR + c] = *(const uint4*)kp;
      if (j0 + 32 < Kk) __builtin_prefetch(kp + 32 * Dd, 0, 0);
    }
    // stage V tile transposed: Vt[d][j] = V[b, j0+j, h*64+d]
    {
      int jj = tid >> 2, dg = (tid & 3) * 16;
      const u16* vrow = Vbuf + baseBK + (size_t)(j0 + jj) * Dd + dg;
      u16 tmp[16];
      *(uint4*)&tmp[0] = *(const uint4*)(vrow);
      *(uint4*)&tmp[8] = *(const uint4*)(vrow + 8);
      for (int e = 0; e < 16; ++e) Vt[(dg + e) * LDS_STR + jj] = tmp[e];
      if (j0 + 32 < Kk) __builtin_prefetch(vrow + 32 * Dd, 0, 0);
    }
    __syncthreads();

    // scores for two 16-key sub-tiles (S = Q x K^T), K fragments from LDS
    v8f s[2];
    for (int jt = 0; jt < 2; ++jt) {
      FragBF kf0, kf1;
      const u16* krow = &Ks[(jt * 16 + n16) * KS_STR];
      load_frag(kf0, krow, half);
      load_frag(kf1, krow + 32, half);
      v8f sa = {};
      sa = wmma_bf16(qf[0], kf0, sa);
      sa = wmma_bf16(qf[1], kf1, sa);
      const bool cj = ctx[(size_t)b * Kk + j0 + jt * 16 + n16] != 0;
      for (int r = 0; r < 8; ++r) {
        float xv = sa[r] * sc2;
        if (((rowctx >> r) & 1u) && !cj) xv = -1.0e30f;
        sa[r] = xv;
      }
      s[jt] = sa;
    }

    // online softmax over the 32-key chunk (DPP16 row reduce across 16 lanes)
    v8f rmx;
    for (int r = 0; r < 8; ++r)
      rmx[r] = red16_max(fmaxf(s[0][r], s[1][r]));

    v8f esc, rsum;
    for (int r = 0; r < 8; ++r) {
      float mn = fmaxf(m8[r], rmx[r]);
      esc[r] = exp2f(m8[r] - mn);
      m8[r] = mn;
      float p0 = exp2f(s[0][r] - mn);
      float p1 = exp2f(s[1][r] - mn);
      s[0][r] = p0; s[1][r] = p1;
      rsum[r] = p0 + p1;
    }
    for (int r = 0; r < 8; ++r)
      rsum[r] = red16_sum(rsum[r]);
    for (int r = 0; r < 8; ++r) l8[r] = l8[r] * esc[r] + rsum[r];
    for (int cf = 0; cf < 4; ++cf)
      for (int r = 0; r < 8; ++r) o[cf][r] *= esc[r];

    // D-layout P -> A-layout via per-wave LDS tile (wave-private; DScnt orders)
    for (int jt = 0; jt < 2; ++jt)
      for (int r = 0; r < 8; ++r)
        myP[(half * 8 + r) * LDS_STR + jt * 16 + n16] = f2bf(s[jt][r]);

    FragBF pf;
    load_frag(pf, &myP[n16 * LDS_STR], half);
    for (int cf = 0; cf < 4; ++cf) {
      FragBF vf;
      load_frag(vf, &Vt[(cf * 16 + n16) * LDS_STR], half);
      o[cf] = wmma_bf16(pf, vf, o[cf]);
    }
  }

  // normalize and store bf16 attention output
  for (int cf = 0; cf < 4; ++cf) {
    const int dcol = h * HDc + cf * 16 + n16;
    for (int r = 0; r < 8; ++r) {
      const int row = q0 + half * 8 + r;
      float y = o[cf][r] / l8[r];
      Ob[(size_t)b * Kk * Dd + (size_t)row * Dd + dcol] = f2bf(y);
    }
  }
}

// ---------------------------------------------------------------------------
// host-side orchestration
// ---------------------------------------------------------------------------
extern "C" void kernel_launch(void* const* d_in, const int* in_sizes, int n_in,
                              void* d_out, int out_size, void* d_ws,
                              size_t ws_size, hipStream_t stream) {
  (void)in_sizes; (void)n_in; (void)out_size; (void)ws_size;

  const float* x  = (const float*)d_in[0];
  const unsigned char* ctx = (const unsigned char*)d_in[1];  // jax bool -> u8
  const float* Wq = (const float*)d_in[2];
  const float* bq = (const float*)d_in[3];
  const float* Wk = (const float*)d_in[4];
  const float* bk = (const float*)d_in[5];
  const float* Wv = (const float*)d_in[6];
  const float* bv = (const float*)d_in[7];
  const float* Wo = (const float*)d_in[8];
  const float* bo = (const float*)d_in[9];
  float* out = (float*)d_out;

  // workspace carve-out (~68 MB total)
  char* ws = (char*)d_ws;
  size_t off = 0;
  auto take = [&](size_t bytes) -> char* {
    char* p = ws + off;
    off += bytes;
    off = (off + 255) & ~(size_t)255;
    return p;
  };
  const size_t xbytes = (size_t)Mm * Dd * sizeof(u16);
  const size_t wbytes = (size_t)Dd * Dd * sizeof(u16);
  u16* xb  = (u16*)take(xbytes);
  u16* wqb = (u16*)take(wbytes);
  u16* wkb = (u16*)take(wbytes);
  u16* wvb = (u16*)take(wbytes);
  u16* wob = (u16*)take(wbytes);
  u16* qb  = (u16*)take(xbytes);
  u16* kb  = (u16*)take(xbytes);
  u16* vb  = (u16*)take(xbytes);
  u16* ab  = (u16*)take(xbytes);

  // stage 0: f32 -> bf16 (vectorized x4)
  cvt_f32_bf16_kernel<<<1024, 256, 0, stream>>>(x,  xb,  Mm * Dd / 4);
  cvt_f32_bf16_kernel<<<256,  256, 0, stream>>>(Wq, wqb, Dd * Dd / 4);
  cvt_f32_bf16_kernel<<<256,  256, 0, stream>>>(Wk, wkb, Dd * Dd / 4);
  cvt_f32_bf16_kernel<<<256,  256, 0, stream>>>(Wv, wvb, Dd * Dd / 4);
  cvt_f32_bf16_kernel<<<256,  256, 0, stream>>>(Wo, wob, Dd * Dd / 4);

  // stage 1: QKV projections
  dim3 ggrid(Mm / 128, Dd / 64);
  gemm_bias_kernel<false><<<ggrid, 256, 0, stream>>>(xb, wqb, bq, qb, nullptr, Mm, Dd, Dd);
  gemm_bias_kernel<false><<<ggrid, 256, 0, stream>>>(xb, wkb, bk, kb, nullptr, Mm, Dd, Dd);
  gemm_bias_kernel<false><<<ggrid, 256, 0, stream>>>(xb, wvb, bv, vb, nullptr, Mm, Dd, Dd);

  // stage 2: masked flash attention
  dim3 agrid(Kk / 64, Hh, Bb);
  flash_attn_kernel<<<agrid, 128, 0, stream>>>(qb, kb, vb, ctx, ab);

  // stage 3: output projection (f32 result)
  gemm_bias_kernel<true><<<ggrid, 256, 0, stream>>>(ab, wob, bo, nullptr, out, Mm, Dd, Dd);
}
